// PairMixerBlock_72902774882787
// MI455X (gfx1250) — matrix operand order repaired and checked by reading, not compiled, not run on hardware
//
#include <hip/hip_runtime.h>
#include <stdint.h>

#define LDIM 384
#define CZ   128
#define LL   (LDIM * LDIM)   // 147456

typedef __attribute__((ext_vector_type(16))) __bf16 v16bf;
typedef __attribute__((ext_vector_type(8)))  float  v8f;

union Frag { v16bf v; uint4 q[2]; };

static __device__ __forceinline__ unsigned short f2bf(float f) {
  union { float f; uint32_t u; } a; a.f = f;
  uint32_t u = a.u;
  u += 0x7FFFu + ((u >> 16) & 1u);            // round-to-nearest-even
  return (unsigned short)(u >> 16);
}
static __device__ __forceinline__ float bf2f(unsigned short h) {
  union { uint32_t u; float f; } a; a.u = ((uint32_t)h) << 16;
  return a.f;
}
// fast sigmoid: v_exp_f32 + v_rcp_f32 (avoid IEEE div_scale/div_fixup chain)
static __device__ __forceinline__ float sigm(float x) {
  return __builtin_amdgcn_rcpf(1.0f + __expf(-x));
}
// fast rsqrt: single v_rsq_f32
static __device__ __forceinline__ float rsq(float x) {
  return __builtin_amdgcn_rsqf(x);
}

static __device__ __forceinline__ v8f zero8() {
  v8f z;
#pragma unroll
  for (int i = 0; i < 8; ++i) z[i] = 0.0f;
  return z;
}

static __device__ __forceinline__ v8f wmma_bf16(Frag a, Frag b, v8f c) {
  return __builtin_amdgcn_wmma_f32_16x16x32_bf16(false, a.v, false, b.v, (short)0, c, false, false);
}

// A fragment (16x32 bf16) from an LDS tile stored [pos][k] (row stride in ushorts).
// ISA layout: lane m=l&15 holds A[m][kb..kb+7] and A[m][kb+16..kb+23], kb=8*(l>>4).
static __device__ __forceinline__ Frag fragA_lds(const unsigned short* lds, int stride,
                                                 int posBase, int lane, int k0) {
  int m  = posBase + (lane & 15);
  int kb = k0 + 8 * (lane >> 4);
  Frag f;
  f.q[0] = *(const uint4*)(lds + m * stride + kb);
  f.q[1] = *(const uint4*)(lds + m * stride + kb + 16);
  return f;
}
// B fragment (32x16 bf16) from storage S[n][k] (k contiguous): lane n=l&15 holds
// K values kb..kb+7 and kb+16..kb+23, kb = k0 + 8*(l>>4).
static __device__ __forceinline__ Frag fragB_rows(const unsigned short* base, int stride,
                                                  int n0, int lane, int k0) {
  const unsigned short* p = base + (size_t)(n0 + (lane & 15)) * stride + k0 + 8 * (lane >> 4);
  Frag f;
  f.q[0] = *(const uint4*)p;
  f.q[1] = *(const uint4*)(p + 16);
  return f;
}

// ---------- weight prep: W[k][n] f32 -> Wt[n][k] bf16 ----------
__global__ void wprep_kernel(const float* __restrict__ src, unsigned short* __restrict__ dst,
                             int K, int N) {
  int o = blockIdx.x * 256 + threadIdx.x;
  if (o < K * N) {
    int n = o / K, k = o % K;
    dst[o] = f2bf(src[(size_t)k * N + n]);
  }
}

// ---------- fused LN + 5 projections (a/b/g) for one tri-mul ----------
// Block = 8x8 position tile. Writes a/b as channel planes plane[c][row][k] (bf16),
// transposed for incoming so the einsum is NT for both blocks; g natural [pos][c].
__global__ __launch_bounds__(256) void proj_kernel(
    const float* __restrict__ zin, const float* __restrict__ mask,
    const float* __restrict__ ln_g, const float* __restrict__ ln_b,
    const unsigned short* __restrict__ Wt_ap, const float* __restrict__ b_ap,
    const unsigned short* __restrict__ Wt_ag, const float* __restrict__ b_ag,
    const unsigned short* __restrict__ Wt_bp, const float* __restrict__ b_bp,
    const unsigned short* __restrict__ Wt_bg, const float* __restrict__ b_bg,
    const unsigned short* __restrict__ Wt_g,  const float* __restrict__ b_g,
    unsigned short* __restrict__ aplane, unsigned short* __restrict__ bplane,
    unsigned short* __restrict__ gbuf, int incoming)
{
  __shared__ unsigned short zn[64 * 136];     // 17.0 KB, padded rows (16B aligned)
  __shared__ unsigned short stage[128 * 72];  // 18.0 KB transpose staging
  const int p0 = blockIdx.x * 8, q0 = blockIdx.y * 8;
  const int tid = threadIdx.x, wave = tid >> 5, lane = tid & 31;

  // --- LayerNorm -> zn (bf16) ---
  float4 lg = *(const float4*)(ln_g + 4 * lane);
  float4 lb = *(const float4*)(ln_b + 4 * lane);
  for (int r = 0; r < 8; ++r) {
    int pos = wave * 8 + r;
    int pp = pos >> 3, qq = pos & 7;
    const float* zp = zin + ((size_t)(p0 + pp) * LDIM + (q0 + qq)) * CZ;
    float4 zv = *(const float4*)(zp + 4 * lane);
    float s  = zv.x + zv.y + zv.z + zv.w;
    float ss = zv.x * zv.x + zv.y * zv.y + zv.z * zv.z + zv.w * zv.w;
#pragma unroll
    for (int off = 16; off > 0; off >>= 1) {
      s  += __shfl_xor(s, off, 32);
      ss += __shfl_xor(ss, off, 32);
    }
    float mu  = s * (1.0f / 128.0f);
    float var = ss * (1.0f / 128.0f) - mu * mu;
    float rs  = rsq(var + 1e-5f);
    ushort4 o;
    o.x = f2bf((zv.x - mu) * rs * lg.x + lb.x);
    o.y = f2bf((zv.y - mu) * rs * lg.y + lb.y);
    o.z = f2bf((zv.z - mu) * rs * lg.z + lb.z);
    o.w = f2bf((zv.w - mu) * rs * lg.w + lb.w);
    *(ushort4*)&zn[pos * 136 + 4 * lane] = o;
  }
  __syncthreads();

  // --- g = sigmoid(zn @ w_g + b_g), natural [pos][c] ---
  for (int t = 0; t < 4; ++t) {
    int j = wave + 8 * t;                    // 32 jobs = 4 msub x 8 nsub
    int msub = j >> 3, n0 = (j & 7) * 16;
    v8f acc = zero8();
#pragma unroll
    for (int ks = 0; ks < 4; ++ks) {
      Frag A = fragA_lds(zn, 136, msub * 16, lane, ks * 32);
      Frag B = fragB_rows(Wt_g, 128, n0, lane, ks * 32);
      acc = wmma_bf16(A, B, acc);
    }
    int c = n0 + (lane & 15);
    float bg = b_g[c];
#pragma unroll
    for (int v = 0; v < 8; ++v) {
      int Me = v + 8 * (lane >> 4);
      int pos = msub * 16 + Me;
      int pp = pos >> 3, qq = pos & 7;
      size_t gp = (size_t)(p0 + pp) * LDIM + (q0 + qq);
      gbuf[gp * CZ + c] = f2bf(sigm(acc[v] + bg));
    }
  }

  // --- a and b: mask * sigmoid(gate) * proj -> channel planes ---
  for (int which = 0; which < 2; ++which) {
    const unsigned short* Wp  = which ? Wt_bp : Wt_ap;
    const unsigned short* Wg2 = which ? Wt_bg : Wt_ag;
    const float* bp_ = which ? b_bp : b_ap;
    const float* bg_ = which ? b_bg : b_ag;
    unsigned short* outp = which ? bplane : aplane;
    for (int t = 0; t < 4; ++t) {
      int j = wave + 8 * t;
      int msub = j >> 3, n0 = (j & 7) * 16;
      v8f accp = zero8(), accg = zero8();
#pragma unroll
      for (int ks = 0; ks < 4; ++ks) {
        Frag A  = fragA_lds(zn, 136, msub * 16, lane, ks * 32);
        Frag Bq = fragB_rows(Wp,  128, n0, lane, ks * 32);
        Frag Bg = fragB_rows(Wg2, 128, n0, lane, ks * 32);
        accp = wmma_bf16(A, Bq, accp);
        accg = wmma_bf16(A, Bg, accg);
      }
      int c = n0 + (lane & 15);
      float bp = bp_[c], bg = bg_[c];
#pragma unroll
      for (int v = 0; v < 8; ++v) {
        int Me = v + 8 * (lane >> 4);
        int pos = msub * 16 + Me;
        int pp = pos >> 3, qq = pos & 7;
        float mv  = mask[(size_t)(p0 + pp) * LDIM + (q0 + qq)];
        float val = mv * sigm(accg[v] + bg) * (accp[v] + bp);
        int row = incoming ? qq : pp;
        int col = incoming ? pp : qq;
        stage[c * 72 + row * 8 + col] = f2bf(val);
      }
    }
    __syncthreads();
    // stage[c][row][0..7] -> plane[c][row_global][col_global..+7] (one b128 each)
    for (int t2 = 0; t2 < 4; ++t2) {
      int idx = tid + 256 * t2;               // 1024 (c,row) pairs
      int c = idx >> 3, row = idx & 7;
      uint4 d = *(const uint4*)&stage[c * 72 + row * 8];
      size_t base = (size_t)c * LL +
                    (size_t)((incoming ? q0 : p0) + row) * LDIM +
                    (size_t)(incoming ? p0 : q0);
      *(uint4*)(outp + base) = d;
    }
    __syncthreads();
  }
}

// ---------- triangle einsum: X[c][u][v] = sum_k Ap[c][u][k] * Bp[c][v][k] ----------
// LDS-free: operand planes (75 MB bf16) are L2-resident on MI455X's 192 MB L2,
// fragments are direct contiguous 16B global loads; global_prefetch hides L2 latency.
__global__ __launch_bounds__(256) void einsum_kernel(
    const unsigned short* __restrict__ Ap, const unsigned short* __restrict__ Bp,
    float* __restrict__ X)
{
  const int u0 = blockIdx.x * 32, v0 = blockIdx.y * 32, c0 = blockIdx.z * 16;
  const int tid = threadIdx.x, wave = tid >> 5, lane = tid & 31;
  v8f acc[2][2][2];
#pragma unroll
  for (int a = 0; a < 2; ++a)
#pragma unroll
    for (int b = 0; b < 2; ++b)
#pragma unroll
      for (int d = 0; d < 2; ++d) acc[a][b][d] = zero8();
  const int mA = lane & 15, kbA = 8 * (lane >> 4);
  const int nB = lane & 15, kbB = 8 * (lane >> 4);
#pragma unroll
  for (int cc = 0; cc < 2; ++cc) {
    const int c = c0 + wave * 2 + cc;
    const unsigned short* Ab = Ap + (size_t)c * LL;
    const unsigned short* Bb = Bp + (size_t)c * LL;
    for (int ks = 0; ks < 12; ++ks) {
      const int k0 = ks * 32;
      Frag A[2], B[2];
#pragma unroll
      for (int s = 0; s < 2; ++s) {
        const unsigned short* pa = Ab + (size_t)(u0 + s * 16 + mA) * LDIM + k0 + kbA;
        A[s].q[0] = *(const uint4*)pa;
        A[s].q[1] = *(const uint4*)(pa + 16);
        const unsigned short* pb = Bb + (size_t)(v0 + s * 16 + nB) * LDIM + k0 + kbB;
        B[s].q[0] = *(const uint4*)pb;
        B[s].q[1] = *(const uint4*)(pb + 16);
        if (ks < 11) {                         // prefetch next k-step (L2 latency hiding)
          __builtin_prefetch(pa + 32, 0, 1);
          __builtin_prefetch(pb + 32, 0, 1);
        }
      }
#pragma unroll
      for (int us = 0; us < 2; ++us)
#pragma unroll
        for (int vs = 0; vs < 2; ++vs)
          acc[cc][us][vs] = wmma_bf16(A[us], B[vs], acc[cc][us][vs]);
    }
  }
#pragma unroll
  for (int cc = 0; cc < 2; ++cc) {
    const int c = c0 + wave * 2 + cc;
    float* Xc = X + (size_t)c * LL;
#pragma unroll
    for (int us = 0; us < 2; ++us)
#pragma unroll
      for (int vs = 0; vs < 2; ++vs)
#pragma unroll
        for (int v = 0; v < 8; ++v) {
          int Me = v + 8 * (lane >> 4), Ne = lane & 15;
          Xc[(size_t)(u0 + us * 16 + Me) * LDIM + (v0 + vs * 16 + Ne)] = acc[cc][us][vs][v];
        }
  }
}

// ---------- fused LN(x) @ w_o + gate + residual ----------
__global__ __launch_bounds__(128) void out_kernel(
    const float* __restrict__ X, const unsigned short* __restrict__ gbuf,
    const float* __restrict__ zres,
    const float* __restrict__ lno_g, const float* __restrict__ lno_b,
    const unsigned short* __restrict__ Wt_o, const float* __restrict__ b_o,
    float* __restrict__ zout)
{
  __shared__ unsigned short zn[128 * 136];    // 34 KB
  const int u0 = blockIdx.x * 16, v0 = blockIdx.y * 16;
  const int tid = threadIdx.x;
  const int ul = tid >> 4, vl = tid & 15;
  const size_t pbase = (size_t)(u0 + ul) * LDIM + (v0 + vl);
  float s = 0.0f, ss = 0.0f;
  for (int c = 0; c < CZ; ++c) {
    float x = X[(size_t)c * LL + pbase];
    s += x; ss += x * x;
  }
  float mu  = s * (1.0f / 128.0f);
  float var = ss * (1.0f / 128.0f) - mu * mu;
  float rs  = rsq(var + 1e-5f);
  for (int c = 0; c < CZ; ++c) {
    float x = X[(size_t)c * LL + pbase];      // second read hits L2 (X fits)
    zn[tid * 136 + c] = f2bf((x - mu) * rs * lno_g[c] + lno_b[c]);
  }
  __syncthreads();
  const int wave = tid >> 5, lane = tid & 31;
  for (int t = 0; t < 16; ++t) {
    int j = wave + 4 * t;                     // 64 jobs = 8 msub x 8 nsub
    int msub = j >> 3, n0 = (j & 7) * 16;
    v8f acc = zero8();
#pragma unroll
    for (int ks = 0; ks < 4; ++ks) {
      Frag A = fragA_lds(zn, 136, msub * 16, lane, ks * 32);
      Frag B = fragB_rows(Wt_o, 128, n0, lane, ks * 32);
      acc = wmma_bf16(A, B, acc);
    }
    int c = n0 + (lane & 15);
    float bo = b_o[c];
#pragma unroll
    for (int v = 0; v < 8; ++v) {
      int Me = v + 8 * (lane >> 4);
      int pos = msub * 16 + Me;
      int gu = u0 + (pos >> 4), gv = v0 + (pos & 15);
      size_t gi = ((size_t)gu * LDIM + gv) * CZ + c;
      zout[gi] = zres[gi] + bf2f(gbuf[gi]) * (acc[v] + bo);
    }
  }
}

// ---------- fused transition: z + relu(LN(z)@W1+b1)@W2+b2 ----------
__global__ __launch_bounds__(256) void trans_kernel(
    const float* __restrict__ zin,
    const float* __restrict__ ln_g, const float* __restrict__ ln_b,
    const unsigned short* __restrict__ Wt1, const float* __restrict__ b1,
    const unsigned short* __restrict__ Wt2, const float* __restrict__ b2,
    float* __restrict__ out)
{
  __shared__ unsigned short xn[32 * 136];     // 8.5 KB
  __shared__ unsigned short hb[32 * 520];     // 32.5 KB hidden (bf16)
  const int m0 = blockIdx.x * 32;
  const int tid = threadIdx.x, wave = tid >> 5, lane = tid & 31;
  float4 lg = *(const float4*)(ln_g + 4 * lane);
  float4 lb = *(const float4*)(ln_b + 4 * lane);
  for (int r = 0; r < 4; ++r) {
    int pos = wave * 4 + r;
    const float* zp = zin + (size_t)(m0 + pos) * CZ;
    float4 zv = *(const float4*)(zp + 4 * lane);
    float s  = zv.x + zv.y + zv.z + zv.w;
    float ss = zv.x * zv.x + zv.y * zv.y + zv.z * zv.z + zv.w * zv.w;
#pragma unroll
    for (int off = 16; off > 0; off >>= 1) {
      s  += __shfl_xor(s, off, 32);
      ss += __shfl_xor(ss, off, 32);
    }
    float mu  = s * (1.0f / 128.0f);
    float var = ss * (1.0f / 128.0f) - mu * mu;
    float rs  = rsq(var + 1e-5f);
    ushort4 o;
    o.x = f2bf((zv.x - mu) * rs * lg.x + lb.x);
    o.y = f2bf((zv.y - mu) * rs * lg.y + lb.y);
    o.z = f2bf((zv.z - mu) * rs * lg.z + lb.z);
    o.w = f2bf((zv.w - mu) * rs * lg.w + lb.w);
    *(ushort4*)&xn[pos * 136 + 4 * lane] = o;
  }
  __syncthreads();
  // hidden = relu(xn @ W1 + b1): M=32, N=512, K=128
  for (int t = 0; t < 8; ++t) {
    int j = wave + 8 * t;                     // 64 jobs = 2 msub x 32 nsub
    int msub = j >> 5, n0 = (j & 31) * 16;
    v8f acc = zero8();
#pragma unroll
    for (int ks = 0; ks < 4; ++ks) {
      Frag A = fragA_lds(xn, 136, msub * 16, lane, ks * 32);
      Frag B = fragB_rows(Wt1, 128, n0, lane, ks * 32);
      acc = wmma_bf16(A, B, acc);
    }
    int c = n0 + (lane & 15);
    float bb = b1[c];
#pragma unroll
    for (int v = 0; v < 8; ++v) {
      int Me = v + 8 * (lane >> 4);
      int pos = msub * 16 + Me;
      float hv = acc[v] + bb;
      hb[pos * 520 + c] = f2bf(hv > 0.0f ? hv : 0.0f);
    }
  }
  __syncthreads();
  // out = zin + hidden @ w2 + b2: M=32, N=128, K=512
  for (int t = 0; t < 2; ++t) {
    int j = wave + 8 * t;                     // 16 jobs = 2 msub x 8 nsub
    int msub = j >> 3, n0 = (j & 7) * 16;
    v8f acc = zero8();
#pragma unroll
    for (int ks = 0; ks < 16; ++ks) {
      Frag A = fragA_lds(hb, 520, msub * 16, lane, ks * 32);
      Frag B = fragB_rows(Wt2, 512, n0, lane, ks * 32);
      acc = wmma_bf16(A, B, acc);
    }
    int c = n0 + (lane & 15);
    float bo = b2[c];
#pragma unroll
    for (int v = 0; v < 8; ++v) {
      int Me = v + 8 * (lane >> 4);
      int pos = msub * 16 + Me;
      size_t gi = (size_t)(m0 + pos) * CZ + c;
      out[gi] = zin[gi] + acc[v] + bo;
    }
  }
}

extern "C" void kernel_launch(void* const* d_in, const int* in_sizes, int n_in,
                              void* d_out, int out_size, void* d_ws, size_t ws_size,
                              hipStream_t stream) {
  (void)in_sizes; (void)n_in; (void)out_size; (void)ws_size;
  const float* z    = (const float*)d_in[0];
  const float* mask = (const float*)d_in[1];
  const float* const* Pin  = (const float* const*)(d_in + 2);   // 16 entries
  const float* const* Pout = (const float* const*)(d_in + 18);  // 16 entries
  const float* const* Pt   = (const float* const*)(d_in + 34);  // 6 entries
  // tri params: 0 ln_g 1 ln_b 2 w_ap 3 b_ap 4 w_ag 5 b_ag 6 w_bp 7 b_bp
  //             8 w_bg 9 b_bg 10 lno_g 11 lno_b 12 w_g 13 b_g 14 w_o 15 b_o
  // trans params: 0 ln_g 1 ln_b 2 w1 3 b1 4 w2 5 b2

  // workspace layout
  unsigned short* WtIn  = (unsigned short*)d_ws;       // 6 x 128x128 bf16
  unsigned short* WtOut = WtIn + 6 * 16384;            // 6 x 128x128 bf16
  unsigned short* Wt1   = WtIn + 12 * 16384;           // 512x128 bf16
  unsigned short* Wt2   = Wt1 + 512 * 128;             // 128x512 bf16
  float* zbuf = (float*)((char*)d_ws + 655360);        // [L,L,C] f32
  float* X    = zbuf + (size_t)LL * CZ;                // planes [C][L][L] f32
  unsigned short* apl = (unsigned short*)(X + (size_t)LL * CZ);
  unsigned short* bpl = apl + (size_t)LL * CZ;
  unsigned short* gbf = bpl + (size_t)LL * CZ;

  auto prep = [&](const float* s, unsigned short* dptr, int K, int N) {
    int n = K * N;
    wprep_kernel<<<(n + 255) / 256, 256, 0, stream>>>(s, dptr, K, N);
  };
  prep(Pin[2],  WtIn + 0 * 16384, 128, 128);   // ap
  prep(Pin[4],  WtIn + 1 * 16384, 128, 128);   // ag
  prep(Pin[6],  WtIn + 2 * 16384, 128, 128);   // bp
  prep(Pin[8],  WtIn + 3 * 16384, 128, 128);   // bg
  prep(Pin[12], WtIn + 4 * 16384, 128, 128);   // g
  prep(Pin[14], WtIn + 5 * 16384, 128, 128);   // o
  prep(Pout[2],  WtOut + 0 * 16384, 128, 128);
  prep(Pout[4],  WtOut + 1 * 16384, 128, 128);
  prep(Pout[6],  WtOut + 2 * 16384, 128, 128);
  prep(Pout[8],  WtOut + 3 * 16384, 128, 128);
  prep(Pout[12], WtOut + 4 * 16384, 128, 128);
  prep(Pout[14], WtOut + 5 * 16384, 128, 128);
  prep(Pt[2], Wt1, 128, 512);
  prep(Pt[4], Wt2, 512, 128);

  dim3 gProj(48, 48), gEin(12, 12, 8), gOut(24, 24);

  // --- tri-mul incoming: z1 = z + block(z) ---
  proj_kernel<<<gProj, 256, 0, stream>>>(z, mask, Pin[0], Pin[1],
      WtIn + 0 * 16384, Pin[3], WtIn + 1 * 16384, Pin[5],
      WtIn + 2 * 16384, Pin[7], WtIn + 3 * 16384, Pin[9],
      WtIn + 4 * 16384, Pin[13], apl, bpl, gbf, 1);
  einsum_kernel<<<gEin, 256, 0, stream>>>(apl, bpl, X);
  out_kernel<<<gOut, 128, 0, stream>>>(X, gbf, z, Pin[10], Pin[11],
      WtIn + 5 * 16384, Pin[15], zbuf);

  // --- tri-mul outgoing: z2 = z1 + block(z1)  (in-place into zbuf) ---
  proj_kernel<<<gProj, 256, 0, stream>>>(zbuf, mask, Pout[0], Pout[1],
      WtOut + 0 * 16384, Pout[3], WtOut + 1 * 16384, Pout[5],
      WtOut + 2 * 16384, Pout[7], WtOut + 3 * 16384, Pout[9],
      WtOut + 4 * 16384, Pout[13], apl, bpl, gbf, 0);
  einsum_kernel<<<gEin, 256, 0, stream>>>(apl, bpl, X);
  out_kernel<<<gOut, 128, 0, stream>>>(X, gbf, zbuf, Pout[10], Pout[11],
      WtOut + 5 * 16384, Pout[15], zbuf);

  // --- transition: out = z2 + MLP(z2) ---
  trans_kernel<<<LL / 32, 256, 0, stream>>>(zbuf, Pt[0], Pt[1], Wt1, Pt[3],
                                            Wt2, Pt[5], (float*)d_out);
}
